// FrontDoorEncoder_81982335746409
// MI455X (gfx1250) — matrix operand order repaired and checked by reading, not compiled, not run on hardware
//
#include <hip/hip_runtime.h>
#include <hip/hip_bf16.h>

// ---------------------------------------------------------------------------
// MI455X / gfx1250 FrontDoorEncoder.
// - All GEMM-like work on v_wmma_f32_16x16x32_bf16 (wave32).
// - k_sim_topk: fused cosine-sim GEMM + top-8; query tile resident in LDS,
//   prototype stream through a triple-buffered async-to-LDS half-panel ring
//   (24 b128 issues per half, ASYNCcnt <= 48).  Steady-state inner loop is
//   pure ds_load + WMMA.
// - k_attn: flash-style online softmax, K/V chunks staged via async-to-LDS.
// - k_gemm: 16x128 tile/wave -> 8 concurrent WMMA accumulator chains.
// ---------------------------------------------------------------------------

#define B_  8
#define L_  256
#define D_  768
#define H_  12
#define DH_ 64
#define N_  16384
#define K_  8
#define ML_ (B_ * L_)        // 2048 local rows
#define MG_ (B_ * L_ * K_)   // 16384 gathered rows
#define NT_ (N_ / 16)        // 1024 prototype panels

typedef __attribute__((ext_vector_type(16))) __bf16 v16bf;
typedef __attribute__((ext_vector_type(8)))  float  v8f;

struct FragB16 {
    union { unsigned u[8]; v16bf v; };
};

__device__ __forceinline__ unsigned short f2b(float f) {
    union { float f; unsigned u; } x; x.f = f;
    unsigned r = x.u + 0x7FFFu + ((x.u >> 16) & 1u);   // round-to-nearest-even
    return (unsigned short)(r >> 16);
}

__device__ __forceinline__ v8f wmma_bf16(const FragB16& a, const FragB16& b, v8f c) {
    return __builtin_amdgcn_wmma_f32_16x16x32_bf16(false, a.v, false, b.v,
                                                   (short)0, c, false, false);
}

// --- CDNA5 async global->LDS (VGLOBAL GLOBAL_LOAD_ASYNC_TO_LDS_B128) --------
__device__ __forceinline__ void async_b128(unsigned lds_off, const void* gaddr) {
    asm volatile("global_load_async_to_lds_b128 %0, %1, off"
                 :: "v"(lds_off), "v"(gaddr) : "memory");
}

template <int N_OUT>
__device__ __forceinline__ void wait_async() {
#if __has_builtin(__builtin_amdgcn_s_wait_asynccnt)
    __builtin_amdgcn_s_wait_asynccnt(N_OUT);
#else
    asm volatile("s_wait_asynccnt %0" :: "n"(N_OUT) : "memory");
#endif
}

// A-matrix fragment, 16x32 bf16, row-major source, contiguous k order used by this file (not the ISA table order):
// lane m = lane&15; VGPR v holds K pair k = (lane>=16?16:0) + v*2
__device__ __forceinline__ FragB16 load_a_rm(const unsigned short* base, int row0,
                                             int ld, int k0, int lane) {
    FragB16 f;
    int m  = lane & 15;
    int kh = (lane & 16) ? 16 : 0;
    const unsigned short* p = base + (size_t)(row0 + m) * ld + k0;
#pragma unroll
    for (int v = 0; v < 8; v++) {
        int k = kh + (v << 1);
        f.u[v] = *(const unsigned*)(p + k);
    }
    return f;
}

// B-matrix fragment, 32x16 bf16, logical B[k][n] = base[(n0+n)*ld + k0+k]
// (K pairs contiguous). Lanes 0-15: K=0..15, lanes 16-31: K=16..31.
__device__ __forceinline__ FragB16 load_bT(const unsigned short* base, int n0,
                                           int ld, int k0, int lane) {
    FragB16 f;
    int n  = lane & 15;
    int kb = (lane & 16) ? 16 : 0;
    const unsigned short* p = base + (size_t)(n0 + n) * ld + k0 + kb;
#pragma unroll
    for (int v = 0; v < 8; v++) f.u[v] = *(const unsigned*)(p + 2 * v);
    return f;
}

// B-matrix fragment from true row-major source B[k][n] = base[(k0r+k)*ld + n0+n]
__device__ __forceinline__ FragB16 load_b_rm(const unsigned short* base, int k0r,
                                             int ld, int n0, int lane) {
    FragB16 f;
    int n  = lane & 15;
    int kb = (lane & 16) ? 16 : 0;
    const unsigned short* p = base + (size_t)(k0r + kb) * ld + n0 + n;
#pragma unroll
    for (int v = 0; v < 8; v++) {
        unsigned lo = p[(size_t)(2 * v) * ld];
        unsigned hi = p[(size_t)(2 * v + 1) * ld];
        f.u[v] = lo | (hi << 16);
    }
    return f;
}

__device__ __forceinline__ float block_reduce_sum256(float v, float* red) {
    int tid = threadIdx.x;
    red[tid] = v;
    __syncthreads();
    for (int s = 128; s > 0; s >>= 1) {
        if (tid < s) red[tid] += red[tid + s];
        __syncthreads();
    }
    float r = red[0];
    __syncthreads();
    return r;
}

// ---------------------------------------------------------------------------
// Elementwise / row kernels
// ---------------------------------------------------------------------------

__global__ __launch_bounds__(256) void k_rownorm(const float* __restrict__ src,
                                                 unsigned short* __restrict__ dst) {
    int r = blockIdx.x, tid = threadIdx.x;
    __shared__ float red[256];
    const float* s = src + (size_t)r * D_;
    float ss = 0.f;
    for (int c = tid; c < D_; c += 256) { float x = s[c]; ss += x * x; }
    ss = block_reduce_sum256(ss, red);
    float inv = rsqrtf(ss);
    unsigned short* o = dst + (size_t)r * D_;
    for (int c = tid; c < D_; c += 256) o[c] = f2b(s[c] * inv);
}

__global__ __launch_bounds__(256) void k_cast(const float* __restrict__ src,
                                              unsigned short* __restrict__ dst, int n) {
    int i = blockIdx.x * 256 + threadIdx.x;
    if (i < n) dst[i] = f2b(src[i]);
}

__global__ __launch_bounds__(256) void k_wcastT(const float* __restrict__ w,
                                                unsigned short* __restrict__ wT) {
    int i = blockIdx.x * 256 + threadIdx.x;
    if (i >= D_ * D_) return;
    int n = i / D_, k = i % D_;
    wT[(size_t)n * D_ + k] = f2b(w[(size_t)k * D_ + n]);
}

// ---------------------------------------------------------------------------
// Fused cosine-similarity GEMM + top-8.  One wave: 16 query rows x all 16384
// prototypes.  Query tile resident in LDS; prototype half-panels (16 rows x
// 384 = 12KB) streamed through a 3-deep async ring with depth-2 prefetch.
// ---------------------------------------------------------------------------
__global__ __launch_bounds__(32) void k_sim_topk(const unsigned short* __restrict__ tnorm,
                                                 const unsigned short* __restrict__ gnorm,
                                                 int* __restrict__ topk) {
    int row0 = blockIdx.x * 16;
    int lane = threadIdx.x & 31;
    __shared__ unsigned short sA[16 * D_];        // 24KB query tile
    __shared__ unsigned short sB[3][16][384];     // 36KB half-panel ring
    __shared__ float s_sc[16][17];

    // stage the 16x768 query tile once
    {
        const unsigned* src = (const unsigned*)(tnorm + (size_t)row0 * D_);
        unsigned* dst = (unsigned*)sA;
        for (int i = lane; i < 16 * D_ / 2; i += 32) dst[i] = src[i];
    }
    __syncthreads();

    // issue half-panel q (panel q/2, half q&1): 24 async b128 issues
    auto issue_q = [&](int q) {
        int ct = q >> 1, h = q & 1;
        const char* gpanel = (const char*)(gnorm + (size_t)ct * 16 * D_);
        unsigned short* buf = &sB[q % 3][0][0];
#pragma unroll
        for (int i = 0; i < 24; i++) {
            int s = i * 32 + lane;                 // 0..767 segment within half
            int r = s / 48, c = s % 48;            // row, 16B segment in row-half
            const char* g = gpanel + (size_t)r * 1536 + h * 768 + c * 16;
            unsigned lds = (unsigned)(size_t)(buf + (size_t)r * 384 + c * 8);
            async_b128(lds, g);
        }
    };

    issue_q(0);
    issue_q(1);

    float tv[8]; int ti[8];
#pragma unroll
    for (int t = 0; t < 8; t++) { tv[t] = -3.0e38f; ti[t] = t; }

    int m0 = (lane >> 4) << 3, n = lane & 15;

#pragma unroll 1
    for (int ct = 0; ct < NT_; ct++) {
        v8f acc[4] = {};
#pragma unroll
        for (int h = 0; h < 2; h++) {
            int q = ct * 2 + h;
            wait_async<24>();                      // q landed; q+1 still flying
            if (q + 2 < 2 * NT_) issue_q(q + 2);   // <=48 outstanding
            const unsigned short* bb = &sB[q % 3][0][0];
#pragma unroll 1
            for (int k0 = 0; k0 < 384; k0 += 64) {
                FragB16 a0 = load_a_rm(sA, 0, D_, h * 384 + k0, lane);
                FragB16 b0 = load_bT(bb, 0, 384, k0, lane);
                acc[2 * h] = wmma_bf16(a0, b0, acc[2 * h]);
                FragB16 a1 = load_a_rm(sA, 0, D_, h * 384 + k0 + 32, lane);
                FragB16 b1 = load_bT(bb, 0, 384, k0 + 32, lane);
                acc[2 * h + 1] = wmma_bf16(a1, b1, acc[2 * h + 1]);
            }
        }

        // ---- combine + top-8 update ----
#pragma unroll
        for (int v = 0; v < 8; v++)
            s_sc[m0 + v][n] = (acc[0][v] + acc[1][v]) + (acc[2][v] + acc[3][v]);
        __syncthreads();
        if (lane < 16) {
            int col0 = ct * 16;
#pragma unroll
            for (int j = 0; j < 16; j++) {
                float sv = s_sc[lane][j];
                int mi = 0; float mv = tv[0];
#pragma unroll
                for (int t = 1; t < 8; t++)
                    if (tv[t] < mv) { mv = tv[t]; mi = t; }
                if (sv > mv) {
#pragma unroll
                    for (int t = 0; t < 8; t++)
                        if (t == mi) { tv[t] = sv; ti[t] = col0 + j; }
                }
            }
        }
        __syncthreads();
    }
    if (lane < 16) {
#pragma unroll
        for (int t = 0; t < 8; t++)
            topk[(size_t)(row0 + lane) * 8 + t] = ti[t];
    }
}

// gather retrieved prototypes as bf16
__global__ __launch_bounds__(256) void k_gather(const float* __restrict__ dict,
                                                const int* __restrict__ topk,
                                                unsigned short* __restrict__ gf) {
    int g  = blockIdx.x;                 // 0..16383
    int b  = g >> 11, lk = g & 2047;
    int l  = lk >> 3, kk = lk & 7;
    int sr = topk[((size_t)(b * L_ + l)) * 8 + kk];
    const float* s = dict + (size_t)sr * D_;
    unsigned short* o = gf + (size_t)g * D_;
    for (int c = threadIdx.x; c < D_; c += 256) o[c] = f2b(s[c]);
}

// ---------------------------------------------------------------------------
// bf16 GEMM: C[M,768] = A[M,768] @ W + bias, WT pre-transposed.
// One wave -> 16x128 tile (8 concurrent WMMA chains per A load).
// ---------------------------------------------------------------------------
__global__ __launch_bounds__(256) void k_gemm(const unsigned short* __restrict__ A,
                                              const unsigned short* __restrict__ WT,
                                              const float* __restrict__ bias,
                                              float* __restrict__ outF,
                                              unsigned short* __restrict__ outB,
                                              int M) {
    int wave = (blockIdx.x * blockDim.x + threadIdx.x) >> 5;
    int lane = threadIdx.x & 31;
    const int ng = D_ / 128;             // 6 column groups
    int mt = wave / ng, nt = wave % ng;
    if (mt >= M / 16) return;            // wave-uniform
    int row0 = mt * 16, col0 = nt * 128;

    v8f acc[8] = {};
#pragma unroll 1
    for (int k0 = 0; k0 < D_; k0 += 32) {
        FragB16 a = load_a_rm(A, row0, D_, k0, lane);
        if (k0 + 32 < D_)
            __builtin_prefetch(A + (size_t)(row0 + (lane & 15)) * D_ + k0 + 32, 0, 1);
#pragma unroll
        for (int c = 0; c < 8; c++) {
            FragB16 b = load_bT(WT, col0 + c * 16, D_, k0, lane);
            acc[c] = wmma_bf16(a, b, acc[c]);
        }
    }
    int m0 = (lane >> 4) << 3, n = lane & 15;
#pragma unroll
    for (int c = 0; c < 8; c++) {
        float bv = bias ? bias[col0 + c * 16 + n] : 0.f;
#pragma unroll
        for (int v = 0; v < 8; v++) {
            float val = acc[c][v] + bv;
            size_t off = (size_t)(row0 + m0 + v) * D_ + col0 + c * 16 + n;
            if (outF) outF[off] = val;
            if (outB) outB[off] = f2b(val);
        }
    }
}

// ---------------------------------------------------------------------------
// Flash attention, one wave per (batch, head, 16-query tile).  K/V 32-key
// head-slices (32x64 bf16) staged by async-to-LDS, double buffered
// (8+8 b128 issues per chunk, ASYNCcnt <= 32).
// ---------------------------------------------------------------------------
__global__ __launch_bounds__(32) void k_attn(const unsigned short* __restrict__ Q,
                                             const unsigned short* __restrict__ Kb,
                                             const unsigned short* __restrict__ Vb,
                                             unsigned short* __restrict__ ctx,
                                             int Sk) {
    int id = blockIdx.x;                  // B*H*(L/16)
    int qt = id & 15;
    int h  = (id >> 4) % H_;
    int b  = id / (H_ * 16);
    int lane  = threadIdx.x & 31;
    int qrow0 = b * L_ + qt * 16;
    int hc    = h * DH_;

    __shared__ unsigned short sK[2][32][64];   // 8KB
    __shared__ unsigned short sV[2][32][64];   // 8KB
    __shared__ float          s_s[16][33];
    __shared__ unsigned short s_p[16][32];
    __shared__ float          s_r[16];

    // stage one 32x64 bf16 slice: lane -> key row, 8 x 16B segments per row
    auto issue_chunk = [&](const unsigned short* src, int krow0, unsigned short* dst) {
#pragma unroll
        for (int i = 0; i < 8; i++) {
            const char* g = (const char*)(src + (size_t)(krow0 + lane) * D_ + hc) + i * 16;
            unsigned lds = (unsigned)(size_t)(dst + (size_t)lane * 64 + i * 8);
            async_b128(lds, g);
        }
    };

    issue_chunk(Kb, b * Sk, &sK[0][0][0]);
    issue_chunk(Vb, b * Sk, &sV[0][0][0]);

    FragB16 q0 = load_a_rm(Q, qrow0, D_, hc,      lane);
    FragB16 q1 = load_a_rm(Q, qrow0, D_, hc + 32, lane);

    v8f o[4] = {};
    float m_i = -3.0e38f, l_i = 0.f;
    int m0 = (lane >> 4) << 3, n = lane & 15;

#pragma unroll 1
    for (int kc = 0; kc < Sk; kc += 32) {
        int p = (kc >> 5) & 1;
        if (kc + 32 < Sk) {
            issue_chunk(Kb, b * Sk + kc + 32, &sK[p ^ 1][0][0]);
            issue_chunk(Vb, b * Sk + kc + 32, &sV[p ^ 1][0][0]);
            wait_async<16>();              // current chunk landed, next in flight
        } else {
            wait_async<0>();
        }
        // S = Q K^T from LDS (4 WMMAs)
#pragma unroll
        for (int ct = 0; ct < 2; ct++) {
            v8f s = {};
            FragB16 kb0 = load_bT(&sK[p][0][0], ct * 16, 64, 0,  lane);
            s = wmma_bf16(q0, kb0, s);
            FragB16 kb1 = load_bT(&sK[p][0][0], ct * 16, 64, 32, lane);
            s = wmma_bf16(q1, kb1, s);
#pragma unroll
            for (int v = 0; v < 8; v++) s_s[m0 + v][ct * 16 + n] = s[v];
        }
        __syncthreads();
        // online softmax: lanes 0-15 each own a query row
        if (lane < 16) {
            int r = lane;
            float sv[32];
            float cmax = m_i;
#pragma unroll
            for (int j = 0; j < 32; j++) {
                sv[j] = s_s[r][j] * 0.125f;          // 1/sqrt(64)
                cmax  = fmaxf(cmax, sv[j]);
            }
            float sf  = __expf(m_i - cmax);
            float sum = 0.f;
#pragma unroll
            for (int j = 0; j < 32; j++) {
                float pv = __expf(sv[j] - cmax);
                sum += pv;
                s_p[r][j] = f2b(pv);
            }
            l_i = l_i * sf + sum;
            m_i = cmax;
            s_r[r] = sf;
        }
        __syncthreads();
        // rescale accumulator, O += P @ V (4 WMMAs, V from LDS)
        float sca[8];
#pragma unroll
        for (int v = 0; v < 8; v++) sca[v] = s_r[m0 + v];
        FragB16 pf = load_a_rm(&s_p[0][0], 0, 32, 0, lane);
#pragma unroll
        for (int c = 0; c < 4; c++) {
#pragma unroll
            for (int v = 0; v < 8; v++) o[c][v] *= sca[v];
            FragB16 vb = load_b_rm(&sV[p][0][0], 0, 64, c * 16, lane);
            o[c] = wmma_bf16(pf, vb, o[c]);
        }
    }
    if (lane < 16) s_r[lane] = 1.0f / l_i;
    __syncthreads();
    float inv[8];
#pragma unroll
    for (int v = 0; v < 8; v++) inv[v] = s_r[m0 + v];
#pragma unroll
    for (int c = 0; c < 4; c++)
#pragma unroll
        for (int v = 0; v < 8; v++)
            ctx[(size_t)(qrow0 + m0 + v) * D_ + hc + c * 16 + n] = f2b(o[c][v] * inv[v]);
}

// LayerNorm(proj + residual) * g + b   (bias bo already folded into proj)
__global__ __launch_bounds__(256) void k_ln_res(const float* __restrict__ proj,
                                                const float* __restrict__ resid,
                                                const float* __restrict__ g,
                                                const float* __restrict__ bb,
                                                float* __restrict__ out) {
    int r = blockIdx.x, tid = threadIdx.x;
    __shared__ float x[D_];
    __shared__ float red[256];
    size_t base = (size_t)r * D_;
    float s = 0.f;
    for (int c = tid; c < D_; c += 256) { float v = proj[base + c] + resid[base + c]; x[c] = v; s += v; }
    s = block_reduce_sum256(s, red);
    float mu = s / (float)D_;
    float vs = 0.f;
    for (int c = tid; c < D_; c += 256) { float d = x[c] - mu; vs += d * d; }
    vs = block_reduce_sum256(vs, red);
    float inv = rsqrtf(vs / (float)D_ + 1e-12f);
    for (int c = tid; c < D_; c += 256) out[base + c] = (x[c] - mu) * inv * g[c] + bb[c];
}

// out = LN(ll+lg); gate = sigmoid(out@aug_w + local@ori_w + b); blend
__global__ __launch_bounds__(256) void k_final(const float* __restrict__ llo,
                                               const float* __restrict__ lgo,
                                               const float* __restrict__ local,
                                               const float* __restrict__ g,
                                               const float* __restrict__ bb,
                                               const float* __restrict__ aug_w,
                                               const float* __restrict__ ori_w,
                                               const float* __restrict__ aug_b,
                                               const float* __restrict__ ori_b,
                                               float* __restrict__ out) {
    int r = blockIdx.x, tid = threadIdx.x;
    __shared__ float x[D_];
    __shared__ float red[256];
    size_t base = (size_t)r * D_;
    float s = 0.f;
    for (int c = tid; c < D_; c += 256) { float v = llo[base + c] + lgo[base + c]; x[c] = v; s += v; }
    s = block_reduce_sum256(s, red);
    float mu = s / (float)D_;
    float vs = 0.f;
    for (int c = tid; c < D_; c += 256) { float d = x[c] - mu; vs += d * d; }
    vs = block_reduce_sum256(vs, red);
    float inv = rsqrtf(vs / (float)D_ + 1e-12f);
    float dot = 0.f;
    for (int c = tid; c < D_; c += 256) {
        float o = (x[c] - mu) * inv * g[c] + bb[c];
        x[c] = o;                                    // own slots only
        dot += o * aug_w[c] + local[base + c] * ori_w[c];
    }
    dot = block_reduce_sum256(dot, red);
    float gate = 1.0f / (1.0f + __expf(-(dot + aug_b[0] + ori_b[0])));
    for (int c = tid; c < D_; c += 256)
        out[base + c] = gate * x[c] + (1.0f - gate) * local[base + c];
}

// ---------------------------------------------------------------------------
// Launch
// ---------------------------------------------------------------------------
extern "C" void kernel_launch(void* const* d_in, const int* in_sizes, int n_in,
                              void* d_out, int out_size, void* d_ws, size_t ws_size,
                              hipStream_t stream) {
    (void)in_sizes; (void)n_in; (void)out_size; (void)ws_size;

    const float* local = (const float*)d_in[0];   // [8,256,768]
    const float* gdict = (const float*)d_in[1];   // [16384,768]
    const float* ll_w[4]; const float* ll_bias[4];
    const float* lg_w[4]; const float* lg_bias[4];
    for (int i = 0; i < 4; i++) {
        ll_w[i]    = (const float*)d_in[2 + i];
        ll_bias[i] = (const float*)d_in[6 + i];
        lg_w[i]    = (const float*)d_in[12 + i];
        lg_bias[i] = (const float*)d_in[16 + i];
    }
    const float* ll_ln_g = (const float*)d_in[10];
    const float* ll_ln_b = (const float*)d_in[11];
    const float* lg_ln_g = (const float*)d_in[20];
    const float* lg_ln_b = (const float*)d_in[21];
    const float* ln_g    = (const float*)d_in[22];
    const float* ln_b    = (const float*)d_in[23];
    const float* aug_w   = (const float*)d_in[24];
    const float* ori_w   = (const float*)d_in[25];
    const float* aug_b   = (const float*)d_in[26];
    const float* ori_b   = (const float*)d_in[27];

    char* ws = (char*)d_ws;
    size_t off = 0;
    auto alloc = [&](size_t bytes) {
        void* p = ws + off;
        off = (off + bytes + 255) & ~(size_t)255;
        return p;
    };
    unsigned short* gnorm  = (unsigned short*)alloc((size_t)N_ * D_ * 2);
    unsigned short* tnorm  = (unsigned short*)alloc((size_t)ML_ * D_ * 2);
    unsigned short* localb = (unsigned short*)alloc((size_t)ML_ * D_ * 2);
    unsigned short* wT[8];
    for (int i = 0; i < 8; i++) wT[i] = (unsigned short*)alloc((size_t)D_ * D_ * 2);
    int*            topk   = (int*)alloc((size_t)ML_ * K_ * 4);
    unsigned short* gfb    = (unsigned short*)alloc((size_t)MG_ * D_ * 2);
    unsigned short* q_ll   = (unsigned short*)alloc((size_t)ML_ * D_ * 2);
    unsigned short* k_ll   = (unsigned short*)alloc((size_t)ML_ * D_ * 2);
    unsigned short* v_ll   = (unsigned short*)alloc((size_t)ML_ * D_ * 2);
    unsigned short* q_lg   = (unsigned short*)alloc((size_t)ML_ * D_ * 2);
    unsigned short* k_lg   = (unsigned short*)alloc((size_t)MG_ * D_ * 2);
    unsigned short* v_lg   = (unsigned short*)alloc((size_t)MG_ * D_ * 2);
    unsigned short* ctx_ll = (unsigned short*)alloc((size_t)ML_ * D_ * 2);
    unsigned short* ctx_lg = (unsigned short*)alloc((size_t)ML_ * D_ * 2);
    float*          p_ll   = (float*)alloc((size_t)ML_ * D_ * 4);
    float*          p_lg   = (float*)alloc((size_t)ML_ * D_ * 4);
    float*          llo    = (float*)alloc((size_t)ML_ * D_ * 4);
    float*          lgo    = (float*)alloc((size_t)ML_ * D_ * 4);

    // 1) normalizations + bf16 casts
    k_rownorm<<<N_,  256, 0, stream>>>(gdict, gnorm);
    k_rownorm<<<ML_, 256, 0, stream>>>(local, tnorm);
    k_cast<<<(ML_ * D_ + 255) / 256, 256, 0, stream>>>(local, localb, ML_ * D_);
    for (int i = 0; i < 4; i++) {
        k_wcastT<<<(D_ * D_ + 255) / 256, 256, 0, stream>>>(ll_w[i], wT[i]);
        k_wcastT<<<(D_ * D_ + 255) / 256, 256, 0, stream>>>(lg_w[i], wT[4 + i]);
    }

    // 2) fused similarity GEMM + top-8, gather
    k_sim_topk<<<ML_ / 16, 32, 0, stream>>>(tnorm, gnorm, topk);
    k_gather<<<MG_, 256, 0, stream>>>(gdict, topk, gfb);

    // 3) projections. waves = (M/16)*6, 8 waves/block
    const int gridL = (ML_ / 16) * 6 / 8;    // 96
    const int gridG = (MG_ / 16) * 6 / 8;    // 768
    k_gemm<<<gridL, 256, 0, stream>>>(localb, wT[0], ll_bias[0], nullptr, q_ll, ML_);
    k_gemm<<<gridL, 256, 0, stream>>>(localb, wT[1], ll_bias[1], nullptr, k_ll, ML_);
    k_gemm<<<gridL, 256, 0, stream>>>(localb, wT[2], ll_bias[2], nullptr, v_ll, ML_);
    k_gemm<<<gridL, 256, 0, stream>>>(localb, wT[4], lg_bias[0], nullptr, q_lg, ML_);
    k_gemm<<<gridG, 256, 0, stream>>>(gfb,    wT[5], lg_bias[1], nullptr, k_lg, MG_);
    k_gemm<<<gridG, 256, 0, stream>>>(gfb,    wT[6], lg_bias[2], nullptr, v_lg, MG_);

    // 4) attention
    const int gridA = B_ * H_ * (L_ / 16);   // 1536
    k_attn<<<gridA, 32, 0, stream>>>(q_ll, k_ll, v_ll, ctx_ll, L_);
    k_attn<<<gridA, 32, 0, stream>>>(q_lg, k_lg, v_lg, ctx_lg, L_ * K_);

    // 5) output projections + residual + LN
    k_gemm<<<gridL, 256, 0, stream>>>(ctx_ll, wT[3], ll_bias[3], p_ll, nullptr, ML_);
    k_gemm<<<gridL, 256, 0, stream>>>(ctx_lg, wT[7], lg_bias[3], p_lg, nullptr, ML_);
    k_ln_res<<<ML_, 256, 0, stream>>>(p_ll, local, ll_ln_g, ll_ln_b, llo);
    k_ln_res<<<ML_, 256, 0, stream>>>(p_lg, local, lg_ln_g, lg_ln_b, lgo);

    // 6) final LN + sigmoid gating
    k_final<<<ML_, 256, 0, stream>>>(llo, lgo, local, ln_g, ln_b,
                                     aug_w, ori_w, aug_b, ori_b, (float*)d_out);
}
